// RelLSTMLayer_89713276879119
// MI455X (gfx1250) — matrix operand order repaired and checked by reading, not compiled, not run on hardware
//
#include <hip/hip_runtime.h>
#include <math.h>

// ---------------------------------------------------------------------------
// RelLSTMLayer for MI455X (gfx1250): bf16 WMMA (f32 accumulate) recurrent GEMMs.
// S=512, B=256, IN=256, BR=128, H=512, G=4H=2048.
// Each wave owns a 16x64 output tile: A fragment reused across 4 WMMAs.
// K segmentation ([h | x | b]) handled by per-segment straight-line loops.
// ---------------------------------------------------------------------------

typedef __attribute__((ext_vector_type(16))) __bf16 v16bf;
typedef __attribute__((ext_vector_type(8)))  float  v8f;

#define S_LEN  512
#define BATCH  256
#define IN_DIM 256
#define BR_DIM 128
#define H_DIM  512
#define G_DIM  2048

union FragBF { v16bf v; uint4 u[2]; };
union FragF  { v8f  v; float f[8];  };

// ---------------- elementwise converts / init ----------------

__global__ __launch_bounds__(256)
void f32_to_bf16_kernel(const float* __restrict__ src, __bf16* __restrict__ dst, long n) {
  long i = (long)blockIdx.x * blockDim.x + threadIdx.x;
  if (i < n) dst[i] = (__bf16)src[i];
}

__global__ __launch_bounds__(256)
void zero_u32_kernel(unsigned* __restrict__ p, long nwords) {
  long i = (long)blockIdx.x * blockDim.x + threadIdx.x;
  if (i < nwords) p[i] = 0u;
}

// Transpose-convert: dst[n*ldDst + colOff + k] = bf16(src[k*N + n])
__global__ __launch_bounds__(256)
void wt_convert_kernel(const float* __restrict__ src, __bf16* __restrict__ dst,
                       int K, int N, int ldDst, int colOff) {
  long idx = (long)blockIdx.x * blockDim.x + threadIdx.x;
  if (idx >= (long)K * N) return;
  int k = (int)(idx / N);
  int n = (int)(idx % N);
  dst[(size_t)n * ldDst + colOff + k] = (__bf16)src[(size_t)k * N + n];
}

// ---------------- per-step recurrent GEMM (bf16 WMMA) ----------------
// C[m][n] = sum_k A[m][k] * Wt[n][k].
// A-fragment layout (16x32 bf16): lanes 0-15 hold K=[0..15] (contiguous, same as B),
// lanes 16-31 hold K=[16..31] of row m = lane%16  -> two 16B loads.
// B-fragment layout (32x16 bf16): lane's n = lane%16 (row of W^T);
// lanes 0-15 hold K=[0..15], lanes 16-31 K=[16..31] -> one 32B span.
// C/D: VGPR i -> m = 8*(lane/16) + i, n = lane%16.

// Process one K-segment: A rows from `arow` (local k in [0,len)),
// weight columns at global-k offset `koff`.
__device__ __forceinline__
void gemm_segment(const __bf16* __restrict__ arow,
                  const __bf16* __restrict__ wr0, const __bf16* __restrict__ wr1,
                  const __bf16* __restrict__ wr2, const __bf16* __restrict__ wr3,
                  int koff, int len, int half, FragF (&acc)[4]) {
  const __bf16* w0 = wr0 + koff + half * 16;
  const __bf16* w1 = wr1 + koff + half * 16;
  const __bf16* w2 = wr2 + koff + half * 16;
  const __bf16* w3 = wr3 + koff + half * 16;
#pragma unroll 2
  for (int kl = 0; kl < len; kl += 32) {
    FragBF A;
    A.u[0] = *(const uint4*)(arow + kl + half * 16);
    A.u[1] = *(const uint4*)(arow + kl + half * 16 + 8);
    FragBF B0, B1, B2, B3;
    B0.u[0] = *(const uint4*)(w0 + kl);
    B0.u[1] = *(const uint4*)(w0 + kl + 8);
    B1.u[0] = *(const uint4*)(w1 + kl);
    B1.u[1] = *(const uint4*)(w1 + kl + 8);
    B2.u[0] = *(const uint4*)(w2 + kl);
    B2.u[1] = *(const uint4*)(w2 + kl + 8);
    B3.u[0] = *(const uint4*)(w3 + kl);
    B3.u[1] = *(const uint4*)(w3 + kl + 8);
    acc[0].v = __builtin_amdgcn_wmma_f32_16x16x32_bf16(false, A.v, false, B0.v,
                                                       (short)0, acc[0].v, false, false);
    acc[1].v = __builtin_amdgcn_wmma_f32_16x16x32_bf16(false, A.v, false, B1.v,
                                                       (short)0, acc[1].v, false, false);
    acc[2].v = __builtin_amdgcn_wmma_f32_16x16x32_bf16(false, A.v, false, B2.v,
                                                       (short)0, acc[2].v, false, false);
    acc[3].v = __builtin_amdgcn_wmma_f32_16x16x32_bf16(false, A.v, false, B3.v,
                                                       (short)0, acc[3].v, false, false);
  }
}

// A is three concatenated K-segments: [a0: len0 | a1: len1 | a2: len2].
// Pass len1=0 / len2=0 for unused segments.
__global__ __launch_bounds__(128)
void lstm_step_gemm(const __bf16* __restrict__ a0, int ld0, int len0,
                    const __bf16* __restrict__ a1, int ld1, int len1,
                    const __bf16* __restrict__ a2, int ld2, int len2,
                    const __bf16* __restrict__ wt, int ktot,
                    float* __restrict__ g) {
  // 512 wave-tiles: 16 M-tiles x 32 N-groups (each group = 4 consecutive 16-col tiles)
  int id   = blockIdx.x * 4 + (threadIdx.x >> 5);
  int tm   = id >> 5;          // [0,16)
  int tg   = id & 31;          // [0,32)
  int lane = threadIdx.x & 31;
  int half = lane >> 4;
  int r    = lane & 15;

  int nbase = tg * 64 + r;
  const __bf16* wr0 = wt + (size_t)(nbase)      * ktot;
  const __bf16* wr1 = wt + (size_t)(nbase + 16) * ktot;
  const __bf16* wr2 = wt + (size_t)(nbase + 32) * ktot;
  const __bf16* wr3 = wt + (size_t)(nbase + 48) * ktot;

  FragF acc[4];
#pragma unroll
  for (int j = 0; j < 4; ++j)
#pragma unroll
    for (int i = 0; i < 8; ++i) acc[j].f[i] = 0.0f;

  int mrow = tm * 16 + r;
  gemm_segment(a0 + (size_t)mrow * ld0, wr0, wr1, wr2, wr3, 0, len0, half, acc);
  if (len1 > 0)
    gemm_segment(a1 + (size_t)mrow * ld1, wr0, wr1, wr2, wr3, len0, len1, half, acc);
  if (len2 > 0)
    gemm_segment(a2 + (size_t)mrow * ld2, wr0, wr1, wr2, wr3, len0 + len1, len2, half, acc);

  float* grow = g + (size_t)(tm * 16 + half * 8) * G_DIM + tg * 64 + r;
#pragma unroll
  for (int j = 0; j < 4; ++j)
#pragma unroll
    for (int i = 0; i < 8; ++i) grow[(size_t)i * G_DIM + j * 16] = acc[j].f[i];
}

// ---------------- LSTM gate pointwise ----------------
__device__ __forceinline__ float sigmoidf_(float x) { return 1.0f / (1.0f + expf(-x)); }

__global__ __launch_bounds__(256)
void lstm_pointwise(const float* __restrict__ g, const float* __restrict__ bias,
                    float* __restrict__ c, float* __restrict__ h,
                    __bf16* __restrict__ hbf, float* __restrict__ hs_out) {
  int idx = blockIdx.x * 256 + threadIdx.x;          // 0 .. B*H-1
  int b = idx >> 9;                                   // /H
  int j = idx & (H_DIM - 1);
  const float* gr = g + (size_t)b * G_DIM;
  float i_ = gr[j]             + bias[j];
  float f_ = gr[H_DIM + j]     + bias[H_DIM + j];
  float gg = gr[2 * H_DIM + j] + bias[2 * H_DIM + j];
  float o_ = gr[3 * H_DIM + j] + bias[3 * H_DIM + j];
  float cn = sigmoidf_(f_) * c[idx] + sigmoidf_(i_) * tanhf(gg);
  float hn = sigmoidf_(o_) * tanhf(cn);
  c[idx]   = cn;
  h[idx]   = hn;
  hbf[idx] = (__bf16)hn;
  if (hs_out) hs_out[idx] = hn;
}

__global__ __launch_bounds__(256)
void copy_states_kernel(const float* __restrict__ h0, const float* __restrict__ h1,
                        const float* __restrict__ c0, const float* __restrict__ c1,
                        float* __restrict__ out) {
  int i = blockIdx.x * 256 + threadIdx.x;
  const int N = BATCH * H_DIM;
  if (i < N) {
    out[i]         = h0[i];
    out[N + i]     = h1[i];
    out[2 * N + i] = c0[i];
    out[3 * N + i] = c1[i];
  }
}

// ---------------------------------------------------------------------------

extern "C" void kernel_launch(void* const* d_in, const int* in_sizes, int n_in,
                              void* d_out, int out_size, void* d_ws, size_t ws_size,
                              hipStream_t stream) {
  (void)in_sizes; (void)n_in; (void)out_size;
  const float* x     = (const float*)d_in[0];
  const float* b     = (const float*)d_in[1];
  const float* W_ih0 = (const float*)d_in[2];
  const float* W_bh0 = (const float*)d_in[3];
  const float* W_hh0 = (const float*)d_in[4];
  const float* bias0 = (const float*)d_in[5];
  const float* W_ih1 = (const float*)d_in[6];
  const float* W_hh1 = (const float*)d_in[7];
  const float* bias1 = (const float*)d_in[8];
  float* out = (float*)d_out;

  char* ws = (char*)d_ws;
  size_t off = 0;
  auto alloc = [&](size_t bytes) -> void* {
    void* p = ws + off;
    off = (off + bytes + 255) & ~(size_t)255;
    return p;
  };

  const long nx = (long)S_LEN * BATCH * IN_DIM;
  const long nb = (long)S_LEN * BATCH * BR_DIM;
  const int  K0 = H_DIM + IN_DIM + BR_DIM;  // 896  (h | x | b)
  const int  K1 = H_DIM + H_DIM;            // 1024 (h1 | h0)
  const int  NS = BATCH * H_DIM;            // 131072 state elems

  __bf16* xbf  = (__bf16*)alloc((size_t)nx * 2);
  __bf16* bbf  = (__bf16*)alloc((size_t)nb * 2);
  __bf16* w0t  = (__bf16*)alloc((size_t)G_DIM * K0 * 2);
  __bf16* w1t  = (__bf16*)alloc((size_t)G_DIM * K1 * 2);
  float*  g0   = (float*)alloc((size_t)BATCH * G_DIM * 4);
  float*  g1   = (float*)alloc((size_t)BATCH * G_DIM * 4);
  __bf16* h0bf = (__bf16*)alloc((size_t)NS * 2);
  __bf16* h1bf = (__bf16*)alloc((size_t)NS * 2);
  float*  h0f  = (float*)alloc((size_t)NS * 4);
  float*  h1f  = (float*)alloc((size_t)NS * 4);
  float*  c0f  = (float*)alloc((size_t)NS * 4);
  float*  c1f  = (float*)alloc((size_t)NS * 4);
  if (off > ws_size) return;  // insufficient scratch (deterministic no-op)

  // --- precompute: bf16 activations + transposed bf16 weights ---
  f32_to_bf16_kernel<<<(unsigned)((nx + 255) / 256), 256, 0, stream>>>(x, xbf, nx);
  f32_to_bf16_kernel<<<(unsigned)((nb + 255) / 256), 256, 0, stream>>>(b, bbf, nb);

  long wg = ((long)H_DIM * G_DIM + 255) / 256;
  wt_convert_kernel<<<(unsigned)wg, 256, 0, stream>>>(W_hh0, w0t, H_DIM, G_DIM, K0, 0);
  wt_convert_kernel<<<(unsigned)(((long)IN_DIM * G_DIM + 255) / 256), 256, 0, stream>>>(
      W_ih0, w0t, IN_DIM, G_DIM, K0, H_DIM);
  wt_convert_kernel<<<(unsigned)(((long)BR_DIM * G_DIM + 255) / 256), 256, 0, stream>>>(
      W_bh0, w0t, BR_DIM, G_DIM, K0, H_DIM + IN_DIM);
  wt_convert_kernel<<<(unsigned)wg, 256, 0, stream>>>(W_hh1, w1t, H_DIM, G_DIM, K1, 0);
  wt_convert_kernel<<<(unsigned)wg, 256, 0, stream>>>(W_ih1, w1t, H_DIM, G_DIM, K1, H_DIM);

  // zero initial h/c state (fp32 + bf16 copies)
  zero_u32_kernel<<<(NS / 2 + 255) / 256, 256, 0, stream>>>((unsigned*)h0bf, NS / 2);
  zero_u32_kernel<<<(NS / 2 + 255) / 256, 256, 0, stream>>>((unsigned*)h1bf, NS / 2);
  zero_u32_kernel<<<(NS + 255) / 256, 256, 0, stream>>>((unsigned*)h0f, NS);
  zero_u32_kernel<<<(NS + 255) / 256, 256, 0, stream>>>((unsigned*)h1f, NS);
  zero_u32_kernel<<<(NS + 255) / 256, 256, 0, stream>>>((unsigned*)c0f, NS);
  zero_u32_kernel<<<(NS + 255) / 256, 256, 0, stream>>>((unsigned*)c1f, NS);

  // --- sequential scan: 2 layers fused per timestep ---
  const int gemm_grid = 512 / 4;  // 512 wave-tiles, 4 waves/WG
  for (int t = 0; t < S_LEN; ++t) {
    const __bf16* xt = xbf + (size_t)t * BATCH * IN_DIM;
    const __bf16* bt = bbf + (size_t)t * BATCH * BR_DIM;

    // layer 0: g0 = [h0 | x_t | b_t] @ [W_hh0; W_ih0; W_bh0]
    lstm_step_gemm<<<gemm_grid, 128, 0, stream>>>(
        h0bf, H_DIM, H_DIM,
        xt, IN_DIM, IN_DIM,
        bt, BR_DIM, BR_DIM,
        w0t, K0, g0);
    lstm_pointwise<<<NS / 256, 256, 0, stream>>>(g0, bias0, c0f, h0f, h0bf, nullptr);

    // layer 1: g1 = [h1 | h0_t] @ [W_hh1; W_ih1]
    lstm_step_gemm<<<gemm_grid, 128, 0, stream>>>(
        h1bf, H_DIM, H_DIM,
        h0bf, H_DIM, H_DIM,
        nullptr, 0, 0,
        w1t, K1, g1);
    lstm_pointwise<<<NS / 256, 256, 0, stream>>>(g1, bias1, c1f, h1f, h1bf,
                                                 out + (size_t)t * NS);
  }

  // final states: h_n = [h0, h1], c_n = [c0, c1] appended after hs1
  copy_states_kernel<<<(NS + 255) / 256, 256, 0, stream>>>(
      h0f, h1f, c0f, c1f, out + (size_t)S_LEN * NS);
}
